// MultiHeadAttention_2843268350153
// MI455X (gfx1250) — compile-verified, matrix operand order checked by reading, not run on hardware
//
#include <hip/hip_runtime.h>
#include <hip/hip_bf16.h>

typedef __attribute__((ext_vector_type(16))) __bf16 v16bf;
typedef __attribute__((ext_vector_type(8)))  float  v8f;
typedef __attribute__((ext_vector_type(8)))  unsigned short ushort8;
typedef __attribute__((ext_vector_type(16))) unsigned short ushort16;
typedef __attribute__((ext_vector_type(4)))  unsigned int v4u;
typedef __attribute__((ext_vector_type(8)))  int v8i_;
typedef __attribute__((ext_vector_type(4)))  int v4i_;

#define D_MODEL 1024
#define NUM_HEADS 16
#define DK 64
#define SEQ 2048
#define BATCH 4
#define MROWS (BATCH * SEQ) /* 8192 */

#define WMMA_BF16(A, B, C) \
  __builtin_amdgcn_wmma_f32_16x16x32_bf16(false, (A), false, (B), (short)0, (C), false, false)

// float -> bf16 bits, round-to-nearest-even
static __device__ __forceinline__ unsigned short bf_bits(float f) {
  unsigned u = __builtin_bit_cast(unsigned, f);
  u += 0x7FFFu + ((u >> 16) & 1u);
  return (unsigned short)(u >> 16);
}

// Load one bf16 A/B fragment: two contiguous 16B chunks at p and p+16 elements.
// Per ISA 16-bit operand layout: lane holds K = {8h..8h+7} U {16+8h..16+8h+7} (h = lane>=16),
// caller passes p already offset by 8*h.
static __device__ __forceinline__ v16bf load_frag(const unsigned short* p) {
  ushort8 lo = *reinterpret_cast<const ushort8*>(p);
  ushort8 hi = *reinterpret_cast<const ushort8*>(p + 16);
  ushort16 w = __builtin_shufflevector(lo, hi, 0,1,2,3,4,5,6,7,8,9,10,11,12,13,14,15);
  return __builtin_bit_cast(v16bf, w);
}

// ---------------------------------------------------------------- TDM helper
// Issue one 2-D tensor_load_to_lds: tile (tile_d0 x tile_d1) of 2-byte elements,
// global row stride stride0 (elements), into LDS at byte offset lds_off (row-major,
// row pitch = tile_d0 elements). Descriptor bitfields per CDNA5 ISA §8.3-8.5.
static __device__ __forceinline__ void tdm_load_2d(
    const unsigned short* gptr, unsigned lds_off,
    unsigned tile_d0, unsigned tile_d1,
    unsigned tens_d0, unsigned tens_d1,
    unsigned long long stride0) {
  unsigned long long ga = (unsigned long long)(size_t)gptr;
  v4u g0;
  g0[0] = 1u;                                            // count=1, user mode
  g0[1] = lds_off;                                       // lds_addr (bytes)
  g0[2] = (unsigned)(ga & 0xFFFFFFFFu);                  // global_addr[31:0]
  g0[3] = (unsigned)((ga >> 32) & 0x1FFFFFFu) | (2u << 30); // addr[56:32] | type=2
  v8i_ g1;
  g1[0] = (int)(1u << 16);                               // data_size=1 (2 bytes)
  g1[1] = (int)((tens_d0 & 0xFFFFu) << 16);              // tensor_dim0[15:0]
  g1[2] = (int)(((tens_d0 >> 16) & 0xFFFFu) | ((tens_d1 & 0xFFFFu) << 16));
  g1[3] = (int)(((tens_d1 >> 16) & 0xFFFFu) | ((tile_d0 & 0xFFFFu) << 16));
  g1[4] = (int)(tile_d1 & 0xFFFFu);                      // tile_dim1, tile_dim2=0
  g1[5] = (int)(stride0 & 0xFFFFFFFFull);                // tensor_dim0_stride[31:0]
  g1[6] = (int)((stride0 >> 32) & 0xFFFFull);            // stride[47:32], dim1_stride=0
  g1[7] = 0;
  v4i_ z4 = {0, 0, 0, 0};
#if __clang_major__ >= 23
  v8i_ z8 = {0, 0, 0, 0, 0, 0, 0, 0};
  __builtin_amdgcn_tensor_load_to_lds(g0, g1, z4, z4, z8, 0);
#else
  __builtin_amdgcn_tensor_load_to_lds(g0, g1, z4, z4, 0);
#endif
}

// ---------------------------------------------------------------- cast x
__global__ __launch_bounds__(256) void cast_x_kernel(
    const float* __restrict__ in, unsigned short* __restrict__ out, int n) {
  int i = blockIdx.x * blockDim.x + threadIdx.x;
  int stride = gridDim.x * blockDim.x;
  for (; i < n; i += stride) out[i] = bf_bits(in[i]);
}

// ------------------------------------------- transpose + cast weights to [N,K] bf16
__global__ __launch_bounds__(256) void transpose_w_kernel(
    const float* __restrict__ Wq, const float* __restrict__ Wk,
    const float* __restrict__ Wv, const float* __restrict__ Wo,
    unsigned short* __restrict__ wt) {
  const int z = blockIdx.y;
  const float* W = (z == 0) ? Wq : (z == 1) ? Wk : (z == 2) ? Wv : Wo;
  unsigned short* dst = wt + (size_t)z * D_MODEL * D_MODEL;
  int tid = blockIdx.x * blockDim.x + threadIdx.x; // 0 .. 1M-1
  int n = tid >> 10;
  int k = tid & (D_MODEL - 1);
  dst[tid] = bf_bits(W[k * D_MODEL + n]); // dst[n*K + k] = W[k*N + n]
}

// ---------------------------------------------------------------- fused QKV GEMM
// y = x @ W + b. Block = 8 waves, block tile 256x64; wave tile 32x64 (2 A frags x
// 4 B frags -> 8 WMMAs per k-step, each LDS B fragment feeds 2 WMMAs).
// Weight tiles (64x128 bf16, 16 KB) TDM-staged into double-buffered LDS by wave 0.
// Q,K written [B,H,T,Dk]; V written transposed [B,H,Dk,T].
__global__ __launch_bounds__(256) void qkv_gemm_kernel(
    const unsigned short* __restrict__ xb,
    const unsigned short* __restrict__ wtq, const unsigned short* __restrict__ wtk,
    const unsigned short* __restrict__ wtv,
    const float* __restrict__ bq, const float* __restrict__ bk, const float* __restrict__ bv,
    unsigned short* __restrict__ Qb, unsigned short* __restrict__ Kb,
    unsigned short* __restrict__ Vtb) {
  __shared__ __align__(16) unsigned short bs[2][64 * 128]; // 32 KB double buffer
  const int wave = threadIdx.x >> 5;
  const int lane = threadIdx.x & 31;
  const int lane16 = lane & 15;
  const int hi = lane >> 4;
  const int mode = blockIdx.z; // 0=Q 1=K 2=V
  const unsigned short* wt = (mode == 0) ? wtq : (mode == 1) ? wtk : wtv;
  const float* bias = (mode == 0) ? bq : (mode == 1) ? bk : bv;
  const int r0 = blockIdx.x * 256 + wave * 32;
  const int n0 = blockIdx.y * 64;

  const unsigned short* wstrip = wt + (size_t)n0 * D_MODEL; // 64 rows x 1024 cols
  if (wave == 0) {
    tdm_load_2d(wstrip, (unsigned)(size_t)&bs[0][0], 128, 64, 128, 64, D_MODEL);
  }

  v8f acc[2][4] = {};
  const unsigned short* arow0 = xb + (size_t)(r0 + lane16) * D_MODEL + 8 * hi;
  const unsigned short* arow1 = xb + (size_t)(r0 + 16 + lane16) * D_MODEL + 8 * hi;
  for (int kk = 0; kk < D_MODEL; kk += 128) {
    const int buf = (kk >> 7) & 1;
    if (wave == 0) {
      if (kk + 128 < D_MODEL) {
        tdm_load_2d(wstrip + kk + 128, (unsigned)(size_t)&bs[buf ^ 1][0],
                    128, 64, 128, 64, D_MODEL);
        __builtin_amdgcn_s_wait_tensorcnt(1); // in-order: oldest (current) tile done
      } else {
        __builtin_amdgcn_s_wait_tensorcnt(0);
      }
    }
    __syncthreads(); // tile `buf` visible to all waves
    const unsigned short* bsp = &bs[buf][0];
#pragma unroll
    for (int k2 = 0; k2 < 128; k2 += 32) {
      v16bf a0 = load_frag(arow0 + kk + k2);
      v16bf a1 = load_frag(arow1 + kk + k2);
#pragma unroll
      for (int j = 0; j < 4; ++j) {
        v16bf b = load_frag(bsp + (size_t)(16 * j + lane16) * 128 + k2 + 8 * hi);
        acc[0][j] = WMMA_BF16(a0, b, acc[0][j]);
        acc[1][j] = WMMA_BF16(a1, b, acc[1][j]);
      }
    }
    __syncthreads(); // all reads done before next TDM overwrites buf^1
  }

#pragma unroll
  for (int s = 0; s < 2; ++s) {
#pragma unroll
    for (int j = 0; j < 4; ++j) {
      const int gn = n0 + 16 * j + lane16;
      const int h = gn >> 6, d = gn & (DK - 1);
      const float bb = bias[gn];
#pragma unroll
      for (int r = 0; r < 8; ++r) {
        const int gm = r0 + 16 * s + r + 8 * hi;
        const int bidx = gm >> 11, t = gm & (SEQ - 1);
        const unsigned short v = bf_bits(acc[s][j][r] + bb);
        if (mode == 2) {
          Vtb[((size_t)(bidx * NUM_HEADS + h) * DK + d) * SEQ + t] = v;
        } else {
          unsigned short* dst = (mode == 0) ? Qb : Kb;
          dst[((size_t)(bidx * NUM_HEADS + h) * SEQ + t) * DK + d] = v;
        }
      }
    }
  }
}

// ---------------------------------------------------------------- flash attention
// Block = 8 waves sharing one (b,h); each wave owns 16 queries. K/V 32-key tiles
// are TDM-staged into double-buffered LDS. Online softmax; P transposed via LDS.
__global__ __launch_bounds__(256) void attn_kernel(
    const unsigned short* __restrict__ Qb, const unsigned short* __restrict__ Kb,
    const unsigned short* __restrict__ Vtb, unsigned short* __restrict__ Ob) {
  __shared__ __align__(16) unsigned short kbuf[2][32 * DK]; // 2 x 4 KB, [key][d]
  __shared__ __align__(16) unsigned short vbuf[2][DK * 32]; // 2 x 4 KB, [d][t]
  __shared__ __align__(16) unsigned short pbuf[8][16 * 32]; // per-wave P transpose
  const int wave = threadIdx.x >> 5;
  const int lane = threadIdx.x & 31;
  const int lane16 = lane & 15;
  const int hi = lane >> 4;
  const int h = blockIdx.y, b = blockIdx.z;
  const int q0 = (blockIdx.x * 8 + wave) * 16;

  const unsigned short* Qh = Qb + (size_t)(b * NUM_HEADS + h) * SEQ * DK;
  const unsigned short* Kh = Kb + (size_t)(b * NUM_HEADS + h) * SEQ * DK;
  const unsigned short* Vh = Vtb + (size_t)(b * NUM_HEADS + h) * DK * SEQ;

  const v16bf qf0 = load_frag(Qh + (size_t)(q0 + lane16) * DK + 8 * hi);
  const v16bf qf1 = load_frag(Qh + (size_t)(q0 + lane16) * DK + 32 + 8 * hi);

  v8f o0 = {}, o1 = {}, o2 = {}, o3 = {};
  float rowM[8], rowL[8];
#pragma unroll
  for (int r = 0; r < 8; ++r) { rowM[r] = -1e30f; rowL[r] = 0.0f; }

  unsigned short* pw = &pbuf[wave][0];

  if (wave == 0) { // preload tile 0
    tdm_load_2d(Kh, (unsigned)(size_t)&kbuf[0][0], DK, 32, DK, 32, DK);
    tdm_load_2d(Vh, (unsigned)(size_t)&vbuf[0][0], 32, DK, 32, DK, SEQ);
  }

  for (int kt = 0; kt < SEQ; kt += 32) {
    const int buf = (kt >> 5) & 1;
    if (wave == 0) {
      if (kt + 32 < SEQ) {
        tdm_load_2d(Kh + (size_t)(kt + 32) * DK, (unsigned)(size_t)&kbuf[buf ^ 1][0],
                    DK, 32, DK, 32, DK);
        tdm_load_2d(Vh + (kt + 32), (unsigned)(size_t)&vbuf[buf ^ 1][0],
                    32, DK, 32, DK, SEQ);
        __builtin_amdgcn_s_wait_tensorcnt(2); // oldest 2 (current tile) done
      } else {
        __builtin_amdgcn_s_wait_tensorcnt(0);
      }
    }
    __syncthreads(); // tile `buf` ready for everyone
    const unsigned short* kbp = &kbuf[buf][0];
    const unsigned short* vbp = &vbuf[buf][0];

    // S = Q @ K^T for 32 keys (two 16x16 f32 frags)
    v8f s0 = {}, s1 = {};
    {
      v16bf kb;
      kb = load_frag(kbp + (size_t)lane16 * DK + 8 * hi);
      s0 = WMMA_BF16(qf0, kb, s0);
      kb = load_frag(kbp + (size_t)lane16 * DK + 32 + 8 * hi);
      s0 = WMMA_BF16(qf1, kb, s0);
      kb = load_frag(kbp + (size_t)(16 + lane16) * DK + 8 * hi);
      s1 = WMMA_BF16(qf0, kb, s1);
      kb = load_frag(kbp + (size_t)(16 + lane16) * DK + 32 + 8 * hi);
      s1 = WMMA_BF16(qf1, kb, s1);
    }
    const float scale = 0.125f; // 1/sqrt(64)
    float p0[8], p1[8], resc[8];
#pragma unroll
    for (int r = 0; r < 8; ++r) {
      float a = s0[r] * scale, c = s1[r] * scale;
      float m = fmaxf(a, c); // row = r + 8*hi; reduce within 16-lane group (same row)
      m = fmaxf(m, __shfl_xor(m, 1));
      m = fmaxf(m, __shfl_xor(m, 2));
      m = fmaxf(m, __shfl_xor(m, 4));
      m = fmaxf(m, __shfl_xor(m, 8));
      const float nm = fmaxf(rowM[r], m);
      resc[r] = __expf(rowM[r] - nm);
      p0[r] = __expf(a - nm);
      p1[r] = __expf(c - nm);
      float sum = p0[r] + p1[r];
      sum += __shfl_xor(sum, 1);
      sum += __shfl_xor(sum, 2);
      sum += __shfl_xor(sum, 4);
      sum += __shfl_xor(sum, 8);
      rowL[r] = rowL[r] * resc[r] + sum;
      rowM[r] = nm;
    }
#pragma unroll
    for (int r = 0; r < 8; ++r) {
      o0[r] *= resc[r]; o1[r] *= resc[r]; o2[r] *= resc[r]; o3[r] *= resc[r];
    }

    // P (C-layout f32) -> LDS -> reload as bf16 A-fragment (16x32).
    // DS ops are in-order per wave; s_wait_dscnt 0 orders stores vs loads.
    asm volatile("" ::: "memory");
#pragma unroll
    for (int r = 0; r < 8; ++r) {
      const int row = r + 8 * hi;
      pw[row * 32 + lane16] = bf_bits(p0[r]);
      pw[row * 32 + 16 + lane16] = bf_bits(p1[r]);
    }
    asm volatile("s_wait_dscnt 0" ::: "memory");
    const v16bf pf = load_frag(pw + lane16 * 32 + 8 * hi);
    asm volatile("s_wait_dscnt 0" ::: "memory");

    // O += P @ V  (vbuf is [d][t] so B-fragments load contiguously)
    v16bf vb;
    vb = load_frag(vbp + (size_t)(0 + lane16) * 32 + 8 * hi);
    o0 = WMMA_BF16(pf, vb, o0);
    vb = load_frag(vbp + (size_t)(16 + lane16) * 32 + 8 * hi);
    o1 = WMMA_BF16(pf, vb, o1);
    vb = load_frag(vbp + (size_t)(32 + lane16) * 32 + 8 * hi);
    o2 = WMMA_BF16(pf, vb, o2);
    vb = load_frag(vbp + (size_t)(48 + lane16) * 32 + 8 * hi);
    o3 = WMMA_BF16(pf, vb, o3);
    __syncthreads(); // reads done before next iteration's TDM overwrites
  }

  // normalize and store attention output bf16 [B*T, D]
#pragma unroll
  for (int r = 0; r < 8; ++r) {
    const float inv = 1.0f / rowL[r];
    const int t = q0 + r + 8 * hi;
    unsigned short* orow = Ob + ((size_t)(b * SEQ + t)) * D_MODEL + h * DK;
    orow[lane16]      = bf_bits(o0[r] * inv);
    orow[16 + lane16] = bf_bits(o1[r] * inv);
    orow[32 + lane16] = bf_bits(o2[r] * inv);
    orow[48 + lane16] = bf_bits(o3[r] * inv);
  }
}

// ---------------------------------------------------------------- output projection
// Same 256x64 block tile / 32x64 wave tile structure as the QKV GEMM; fp32 output.
__global__ __launch_bounds__(256) void out_gemm_kernel(
    const unsigned short* __restrict__ Ab, const unsigned short* __restrict__ wto,
    const float* __restrict__ bo, float* __restrict__ out) {
  __shared__ __align__(16) unsigned short bs[2][64 * 128];
  const int wave = threadIdx.x >> 5;
  const int lane = threadIdx.x & 31;
  const int lane16 = lane & 15;
  const int hi = lane >> 4;
  const int r0 = blockIdx.x * 256 + wave * 32;
  const int n0 = blockIdx.y * 64;

  const unsigned short* wstrip = wto + (size_t)n0 * D_MODEL;
  if (wave == 0) {
    tdm_load_2d(wstrip, (unsigned)(size_t)&bs[0][0], 128, 64, 128, 64, D_MODEL);
  }

  v8f acc[2][4] = {};
  const unsigned short* arow0 = Ab + (size_t)(r0 + lane16) * D_MODEL + 8 * hi;
  const unsigned short* arow1 = Ab + (size_t)(r0 + 16 + lane16) * D_MODEL + 8 * hi;
  for (int kk = 0; kk < D_MODEL; kk += 128) {
    const int buf = (kk >> 7) & 1;
    if (wave == 0) {
      if (kk + 128 < D_MODEL) {
        tdm_load_2d(wstrip + kk + 128, (unsigned)(size_t)&bs[buf ^ 1][0],
                    128, 64, 128, 64, D_MODEL);
        __builtin_amdgcn_s_wait_tensorcnt(1);
      } else {
        __builtin_amdgcn_s_wait_tensorcnt(0);
      }
    }
    __syncthreads();
    const unsigned short* bsp = &bs[buf][0];
#pragma unroll
    for (int k2 = 0; k2 < 128; k2 += 32) {
      v16bf a0 = load_frag(arow0 + kk + k2);
      v16bf a1 = load_frag(arow1 + kk + k2);
#pragma unroll
      for (int j = 0; j < 4; ++j) {
        v16bf b = load_frag(bsp + (size_t)(16 * j + lane16) * 128 + k2 + 8 * hi);
        acc[0][j] = WMMA_BF16(a0, b, acc[0][j]);
        acc[1][j] = WMMA_BF16(a1, b, acc[1][j]);
      }
    }
    __syncthreads();
  }
#pragma unroll
  for (int s = 0; s < 2; ++s) {
#pragma unroll
    for (int j = 0; j < 4; ++j) {
      const int gn = n0 + 16 * j + lane16;
      const float bb = bo[gn];
#pragma unroll
      for (int r = 0; r < 8; ++r) {
        const int gm = r0 + 16 * s + r + 8 * hi;
        out[(size_t)gm * D_MODEL + gn] = acc[s][j][r] + bb;
      }
    }
  }
}

// ---------------------------------------------------------------- launch
extern "C" void kernel_launch(void* const* d_in, const int* in_sizes, int n_in,
                              void* d_out, int out_size, void* d_ws, size_t ws_size,
                              hipStream_t stream) {
  const float* x  = (const float*)d_in[0];
  const float* Wq = (const float*)d_in[1];
  const float* bq = (const float*)d_in[2];
  const float* Wk = (const float*)d_in[3];
  const float* bk = (const float*)d_in[4];
  const float* Wv = (const float*)d_in[5];
  const float* bv = (const float*)d_in[6];
  const float* Wo = (const float*)d_in[7];
  const float* bo = (const float*)d_in[8];
  float* out = (float*)d_out;

  char* ws = (char*)d_ws;
  const size_t MB = 1024u * 1024u;
  unsigned short* xb  = (unsigned short*)(ws);            // 16 MB: x in bf16 [8192,1024]
  unsigned short* wt  = (unsigned short*)(ws + 16 * MB);  // 8 MB: 4 transposed weights [N,K]
  unsigned short* wtq = wt;
  unsigned short* wtk = wt + 1 * (size_t)D_MODEL * D_MODEL;
  unsigned short* wtv = wt + 2 * (size_t)D_MODEL * D_MODEL;
  unsigned short* wto = wt + 3 * (size_t)D_MODEL * D_MODEL;
  unsigned short* Qb  = (unsigned short*)(ws + 24 * MB);  // 16 MB [B,H,T,Dk]
  unsigned short* Kb  = (unsigned short*)(ws + 40 * MB);  // 16 MB [B,H,T,Dk]
  unsigned short* Vtb = (unsigned short*)(ws + 56 * MB);  // 16 MB [B,H,Dk,T]
  unsigned short* Ob  = (unsigned short*)(ws + 72 * MB);  // 16 MB attn out bf16 [8192,1024]

  cast_x_kernel<<<4096, 256, 0, stream>>>(x, xb, MROWS * D_MODEL);
  transpose_w_kernel<<<dim3(4096, 4, 1), 256, 0, stream>>>(Wq, Wk, Wv, Wo, wt);
  qkv_gemm_kernel<<<dim3(MROWS / 256, D_MODEL / 64, 3), 256, 0, stream>>>(
      xb, wtq, wtk, wtv, bq, bk, bv, Qb, Kb, Vtb);
  attn_kernel<<<dim3(SEQ / 128, NUM_HEADS, BATCH), 256, 0, stream>>>(Qb, Kb, Vtb, Ob);
  out_gemm_kernel<<<dim3(MROWS / 256, D_MODEL / 64), 256, 0, stream>>>(Ob, wto, bo, out);
}
